// SymplecticLeapfrogLayer_2671469658242
// MI455X (gfx1250) — 8-bit matrix operand order repaired and checked by reading, compiled, not run on hardware
//
#include <hip/hip_runtime.h>
#include <hip/hip_bf16.h>


typedef __attribute__((ext_vector_type(8)))  float    v8f;
typedef __attribute__((ext_vector_type(16))) int      v16i;
typedef __attribute__((ext_vector_type(2)))  _Float16 h2v;
typedef int async_v4i __attribute__((vector_size(16)));

#define LF_DT 0.1f

__device__ __forceinline__ unsigned char f32_to_e4m3_sw(float x) {
    if (x != x) return 0x7F;
    unsigned int u   = __float_as_uint(x);
    unsigned int sgn = (u >> 31) << 7;
    float a = fabsf(x);
    if (a >= 448.0f)        return (unsigned char)(sgn | 0x7E);
    if (a < 0.0009765625f)  return (unsigned char)sgn;
    int e;
    float m = frexpf(a, &e);
    int E = e + 6;
    if (E >= 1) {
        int q = (int)rintf(m * 16.0f);
        if (q == 16) { q = 8; E += 1; }
        if (E >= 16) return (unsigned char)(sgn | 0x7E);
        return (unsigned char)(sgn | (E << 3) | (q - 8));
    } else {
        int q = (int)rintf(ldexpf(a, 9));
        if (q >= 8) return (unsigned char)(sgn | 0x08);
        return (unsigned char)(sgn | q);
    }
}

__device__ __forceinline__ unsigned char cvt_e4m3(float x) {
#if defined(__AMDGCN__) && __has_builtin(__builtin_amdgcn_cvt_pk_fp8_f32)
    return (unsigned char)(__builtin_amdgcn_cvt_pk_fp8_f32(x, x, 0, false) & 0xff);
#else
    return f32_to_e4m3_sw(x);
#endif
}

__device__ __forceinline__ float fast_tanh(float x) {
#if defined(__AMDGCN__) && __has_builtin(__builtin_amdgcn_tanhf)
    return __builtin_amdgcn_tanhf(x);
#else
    return tanhf(x);
#endif
}

__global__ void pack_weights_fp8(const float* __restrict__ W1,
                                 const float* __restrict__ W2,
                                 unsigned char* __restrict__ ws) {
    int tid = blockIdx.x * blockDim.x + threadIdx.x;
    if (tid >= 65536) return;
    const int j    = tid & 63;
    const int lane = (tid >> 6) & 31;
    const int nt   = (tid >> 11) & 15;
    const int kt   = (tid >> 15) & 1;
    const int hi   = lane >> 4;
    const int c    = j >> 4;
    const int t    = j & 15;
    const int k = kt * 128 + c * 32 + ((t >> 3) & 1) * 16 + hi * 8 + (t & 7);
    const int n = nt * 16 + (lane & 15);
    ws[tid]          = f32_to_e4m3_sw(W1[k * 256 + n]);
    ws[65536 + tid]  = f32_to_e4m3_sw(W1[n * 256 + k]);
    ws[131072 + tid] = f32_to_e4m3_sw(W2[k * 256 + n]);
    ws[196608 + tid] = f32_to_e4m3_sw(W2[n * 256 + k]);
}

__device__ __forceinline__ void stage_col(unsigned char* st, int off,
                                          const float* v8) {
#pragma unroll
    for (int r = 0; r < 8; ++r)
        st[off + r * 64] = cvt_e4m3(v8[r]);
}

__device__ __forceinline__ v16i bfrag(const unsigned char* W, int kt, int nt,
                                      int lane) {
    return *(const v16i*)(W + (((kt * 16 + nt) * 32) + lane) * 64);
}

__device__ __forceinline__ v8f wmma8(v16i a, v16i b, v8f c) {
    return __builtin_amdgcn_wmma_f32_16x16x128_fp8_fp8(a, b, (short)0, c,
                                                       false, false);
}

template <bool FULL>
__device__ __forceinline__ void dH_update(
    float (&s)[16][8], h2v (&h1h)[16][4],
    const unsigned char* W1f, const unsigned char* W1b,
    const unsigned char* W2f, const unsigned char* W2b,
    unsigned char* st, int lane, int nn, const int (&stoff)[16],
    const float* b1s, const float* b2s, const float* wos) {
#pragma unroll
    for (int nt = 0; nt < 16; ++nt) stage_col(st, stoff[nt], s[nt]);
    v16i a0 = *(const v16i*)(st + lane * 64);
    v16i a1 = *(const v16i*)(st + 2048 + lane * 64);

#pragma unroll
    for (int nt = 0; nt < 16; ++nt) {
        const float bb = b1s[nt * 16 + nn];
        v8f acc;
#pragma unroll
        for (int r = 0; r < 8; ++r) acc[r] = bb;
        acc = wmma8(a0, bfrag(W1f, 0, nt, lane), acc);
        acc = wmma8(a1, bfrag(W1f, 1, nt, lane), acc);
        float hv[8];
#pragma unroll
        for (int r = 0; r < 8; ++r) hv[r] = fast_tanh(acc[r]);
#pragma unroll
        for (int r = 0; r < 4; ++r) {
            h2v t; t.x = (_Float16)hv[2 * r]; t.y = (_Float16)hv[2 * r + 1];
            h1h[nt][r] = t;
        }
        stage_col(st, stoff[nt], hv);
    }
    a0 = *(const v16i*)(st + lane * 64);
    a1 = *(const v16i*)(st + 2048 + lane * 64);

#pragma unroll
    for (int nt = 0; nt < 16; ++nt) {
        const float bb = b2s[nt * 16 + nn];
        const float wo = wos[nt * 16 + nn];
        v8f acc;
#pragma unroll
        for (int r = 0; r < 8; ++r) acc[r] = bb;
        acc = wmma8(a0, bfrag(W2f, 0, nt, lane), acc);
        acc = wmma8(a1, bfrag(W2f, 1, nt, lane), acc);
        float gv[8];
#pragma unroll
        for (int r = 0; r < 8; ++r) {
            float h2 = fast_tanh(acc[r]);
            gv[r] = (1.0f - h2 * h2) * wo;
        }
        stage_col(st, stoff[nt], gv);
    }
    a0 = *(const v16i*)(st + lane * 64);
    a1 = *(const v16i*)(st + 2048 + lane * 64);

#pragma unroll
    for (int nt = 0; nt < 16; ++nt) {
        v8f acc = {};
        acc = wmma8(a0, bfrag(W2b, 0, nt, lane), acc);
        acc = wmma8(a1, bfrag(W2b, 1, nt, lane), acc);
        float gv[8];
#pragma unroll
        for (int r = 0; r < 8; ++r) {
            float h1 = (float)((r & 1) ? h1h[nt][r >> 1].y : h1h[nt][r >> 1].x);
            gv[r] = (1.0f - h1 * h1) * acc[r];
        }
        stage_col(st, stoff[nt], gv);
    }
    a0 = *(const v16i*)(st + lane * 64);
    a1 = *(const v16i*)(st + 2048 + lane * 64);

    constexpr int NT = FULL ? 16 : 8;
#pragma unroll
    for (int nt = 0; nt < NT; ++nt) {
        v8f acc = {};
        acc = wmma8(a0, bfrag(W1b, 0, nt, lane), acc);
        acc = wmma8(a1, bfrag(W1b, 1, nt, lane), acc);
        if (nt < 8) {
#pragma unroll
            for (int r = 0; r < 8; ++r) s[8 + nt][r] -= 0.5f * LF_DT * acc[r];
        } else {
#pragma unroll
            for (int r = 0; r < 8; ++r) s[nt - 8][r] += LF_DT * acc[r];
        }
    }
}

__global__ __launch_bounds__(256) void leapfrog_fp8_wmma(
    const float* __restrict__ x, const float* __restrict__ b1,
    const float* __restrict__ b2, const float* __restrict__ Wout,
    unsigned char* wpack, float* __restrict__ out) {
    extern __shared__ unsigned char smem[];
    unsigned char* W1f = smem;
    unsigned char* W1b = smem + 65536;
    unsigned char* W2f = smem + 131072;
    unsigned char* W2b = smem + 196608;
    unsigned char* stg = smem + 262144;
    float* b1s = (float*)(smem + 294912);
    float* b2s = (float*)(smem + 295936);
    float* wos = (float*)(smem + 296960);

    const int tid  = threadIdx.x;
    const int lane = tid & 31;
    const int wave = tid >> 5;
    const int hi   = lane >> 4;
    const int nn   = lane & 15;

    b1s[tid] = b1[tid];
    b2s[tid] = b2[tid];
    wos[tid] = Wout[tid];

#if defined(__AMDGCN__) && __has_builtin(__builtin_amdgcn_global_load_async_to_lds_b128)
    {
        __attribute__((address_space(1))) char* gsrc =
            (__attribute__((address_space(1))) char*)(void*)wpack;
        __attribute__((address_space(3))) char* ldst =
            (__attribute__((address_space(3))) char*)smem;
#pragma unroll 4
        for (int i = tid; i < 16384; i += 256)
            __builtin_amdgcn_global_load_async_to_lds_b128(
                (__attribute__((address_space(1))) async_v4i*)(gsrc + i * 16),
                (__attribute__((address_space(3))) async_v4i*)(ldst + i * 16),
                0, 0);
        asm volatile("s_wait_asynccnt 0x0" ::: "memory");
    }
#else
    {
        const uint4* src = (const uint4*)wpack;
        uint4* dst = (uint4*)smem;
#pragma unroll 4
        for (int i = tid; i < 16384; i += 256) dst[i] = src[i];
    }
#endif
    __syncthreads();

    unsigned char* st = stg + wave * 4096;
    const int rowbase = (blockIdx.x * 8 + wave) * 16;

    int stoff[16];
#pragma unroll
    for (int nt = 0; nt < 16; ++nt) {
        const int k   = nt * 16 + nn;
        const int kt  = k >> 7;
        const int kl  = k & 127;
        const int hia = (kl >> 3) & 1;
        const int vv  = ((kl >> 2) & 1) | (((kl >> 4) & 1) << 1) |
                        (((kl >> 5) & 1) << 2) | (((kl >> 6) & 1) << 3);
        stoff[nt] = kt * 2048 + (hi * 8 + 16 * hia) * 64 + vv * 4 + (kl & 3);
    }

    float s[16][8];
    h2v   h1h[16][4];
#pragma unroll
    for (int nt = 0; nt < 8; ++nt) {
        const int f = nt * 16 + nn;
#pragma unroll
        for (int r = 0; r < 8; ++r) {
            const long b = rowbase + hi * 8 + r;
            float x15 = x[b * 2048 + 15 * 128 + f];
            float x14 = x[b * 2048 + 14 * 128 + f];
            s[nt][r]     = x15;
            s[8 + nt][r] = x15 - x14;
        }
    }

#pragma unroll 1
    for (int step = 0; step < 3; ++step) {
        dH_update<true >(s, h1h, W1f, W1b, W2f, W2b, st, lane, nn, stoff,
                         b1s, b2s, wos);
        dH_update<false>(s, h1h, W1f, W1b, W2f, W2b, st, lane, nn, stoff,
                         b1s, b2s, wos);
    }

#pragma unroll
    for (int nt = 0; nt < 16; ++nt) {
#pragma unroll
        for (int r = 0; r < 8; ++r) {
            const long b = rowbase + hi * 8 + r;
            out[b * 256 + nt * 16 + nn] = s[nt][r];
        }
    }
}

extern "C" void kernel_launch(void* const* d_in, const int* in_sizes, int n_in,
                              void* d_out, int out_size, void* d_ws,
                              size_t ws_size, hipStream_t stream) {
    (void)in_sizes; (void)n_in; (void)out_size; (void)ws_size;
    const float* x    = (const float*)d_in[0];
    const float* W1   = (const float*)d_in[1];
    const float* b1   = (const float*)d_in[2];
    const float* W2   = (const float*)d_in[3];
    const float* b2   = (const float*)d_in[4];
    const float* Wout = (const float*)d_in[5];
    unsigned char* wpack = (unsigned char*)d_ws;
    float* out = (float*)d_out;

    (void)hipFuncSetAttribute((const void*)leapfrog_fp8_wmma,
                              hipFuncAttributeMaxDynamicSharedMemorySize,
                              297984);

    pack_weights_fp8<<<dim3(256), dim3(256), 0, stream>>>(W1, W2, wpack);
    leapfrog_fp8_wmma<<<dim3(512), dim3(256), 297984, stream>>>(
        x, b1, b2, Wout, wpack, out);
}
